// VariationalEncoder_60593398612125
// MI455X (gfx1250) — matrix operand order repaired and checked by reading, not compiled, not run on hardware
//
#include <hip/hip_runtime.h>
#include <hip/hip_bf16.h>

typedef __attribute__((ext_vector_type(16))) _Float16 v16h;
typedef __attribute__((ext_vector_type(8)))  _Float16 v8h;
typedef __attribute__((ext_vector_type(8)))  float    v8f;

__device__ __forceinline__ v16h cat8(v8h lo, v8h hi) {
  return __builtin_shufflevector(lo, hi, 0,1,2,3,4,5,6,7,8,9,10,11,12,13,14,15);
}

__device__ __forceinline__ v8f wmma_f16(v16h a, v16h b, v8f c) {
  // D = A(16x32 f16) * B(32x16 f16) + C(16x16 f32)
  return __builtin_amdgcn_wmma_f32_16x16x32_f16(false, a, false, b, (short)0, c, false, false);
}

// ---------------------------------------------------------------------------
// Elementwise helpers
// ---------------------------------------------------------------------------
__global__ void fill_f32(float* __restrict__ p, float v, size_t n) {
  size_t stride = (size_t)gridDim.x * blockDim.x;
  for (size_t i = (size_t)blockIdx.x * blockDim.x + threadIdx.x; i < n; i += stride)
    p[i] = v;
}

__global__ void f32_to_f16(const float* __restrict__ in, _Float16* __restrict__ out, size_t n) {
  size_t stride = (size_t)gridDim.x * blockDim.x;
  for (size_t i = (size_t)blockIdx.x * blockDim.x + threadIdx.x; i < n; i += stride)
    out[i] = (_Float16)in[i];
}

__global__ void rsqrt_inplace(float* __restrict__ p, size_t n) {
  size_t stride = (size_t)gridDim.x * blockDim.x;
  for (size_t i = (size_t)blockIdx.x * blockDim.x + threadIdx.x; i < n; i += stride) {
    float d = p[i];
    p[i] = d > 0.0f ? rsqrtf(d) : 0.0f;
  }
}

__global__ void relu_bias_to_f16(const float* __restrict__ acc, const float* __restrict__ bias,
                                 _Float16* __restrict__ out, size_t total, int fmask) {
  size_t stride = (size_t)gridDim.x * blockDim.x;
  for (size_t i = (size_t)blockIdx.x * blockDim.x + threadIdx.x; i < total; i += stride) {
    float v = acc[i] + bias[i & (size_t)fmask];
    out[i] = (_Float16)fmaxf(v, 0.0f);
  }
}

__global__ void bias_add_inplace(float* __restrict__ out, const float* __restrict__ bias,
                                 size_t total, int fmask) {
  size_t stride = (size_t)gridDim.x * blockDim.x;
  for (size_t i = (size_t)blockIdx.x * blockDim.x + threadIdx.x; i < total; i += stride)
    out[i] += bias[i & (size_t)fmask];
}

// ---------------------------------------------------------------------------
// Degree (with self-loops) and scatter aggregation
// ---------------------------------------------------------------------------
__global__ void deg_accum(const long long* __restrict__ dst, float* __restrict__ deg, int E) {
  int stride = gridDim.x * blockDim.x;
  for (int e = blockIdx.x * blockDim.x + threadIdx.x; e < E; e += stride)
    unsafeAtomicAdd(&deg[(int)dst[e]], 1.0f);
}

// out[dst] += dinv[src]*dinv[dst] * xw[src]  over E real edges + Nn self loops.
// F = 1<<logF channels; thread i handles (edge i>>logF, channel i&(F-1)).
__global__ void agg_scatter(const float* __restrict__ xw,
                            const long long* __restrict__ src,
                            const long long* __restrict__ dst,
                            const float* __restrict__ dinv,
                            float* __restrict__ out, int E, int Nn, int logF) {
  const int fmask = (1 << logF) - 1;
  const size_t total = ((size_t)E + (size_t)Nn) << logF;
  const size_t stride = (size_t)gridDim.x * blockDim.x;
  for (size_t i = (size_t)blockIdx.x * blockDim.x + threadIdx.x; i < total; i += stride) {
    size_t e = i >> logF;
    int c = (int)(i & (size_t)fmask);
    int s, d;
    if (e < (size_t)E) {
      s = (int)src[e];
      d = (int)dst[e];
    } else {
      s = d = (int)(e - (size_t)E);
    }
    float w = dinv[s] * dinv[d];
    unsafeAtomicAdd(out + (((size_t)d << logF) + c), xw[((size_t)s << logF) + c] * w);
  }
}

// ---------------------------------------------------------------------------
// Pack B (KxN f32 row-major) into WMMA fragment tiles:
// tile t = kt*(N/16)+nt holds 32x16 f16; lane L owns column nt*16+(L&15),
// K = kt*32 + (L>>4)*8 + (j&7) + (j>>3)*16, stored contiguously (16 f16 = 32B per lane).
// ---------------------------------------------------------------------------
__global__ void pack_b_f16(const float* __restrict__ B, _Float16* __restrict__ Bp, int K, int N) {
  int total = K * N;
  int ntile16 = N >> 4;
  int stride = gridDim.x * blockDim.x;
  for (int o = blockIdx.x * blockDim.x + threadIdx.x; o < total; o += stride) {
    int t   = o >> 9;       // /512 elements per tile
    int rem = o & 511;
    int L   = rem >> 4;
    int j   = rem & 15;
    int kt  = t / ntile16;
    int nt  = t - kt * ntile16;
    int k   = (kt << 5) + ((L >> 4) << 3) + (j & 7) + ((j >> 3) << 4);
    int n   = (nt << 4) + (L & 15);
    Bp[o] = (_Float16)B[(size_t)k * N + n];
  }
}

// ---------------------------------------------------------------------------
// WMMA GEMM: C[MxN f32] = A[MxK f16 row-major] * Bp[packed f16]
// 256 threads = 8 waves; wave w -> rows [ (bx*8+w)*16 , +16 ), cols [by*64, +64).
// M % 16 == 0, N % 64 == 0, K % 32 == 0 (holds for all three GEMMs here).
// ---------------------------------------------------------------------------
__global__ void __launch_bounds__(256)
gemm16_wmma(const _Float16* __restrict__ A, const _Float16* __restrict__ Bp,
            float* __restrict__ C, int M, int N, int K) {
  const int lane = threadIdx.x & 31;
  const int wave = threadIdx.x >> 5;
  const int m0 = (blockIdx.x * 8 + wave) * 16;
  if (m0 >= M) return;                       // wave-uniform: EXEC stays all-ones
  const int n0 = blockIdx.y * 64;
  const int half = lane >> 4;                // which 16-lane group
  const int ml = lane & 15;
  const int ntile16 = N >> 4;

  // A fragment: lane holds row m0+ml, K = k0 + half*8 + {0..7} and +16..23
  const _Float16* arow = A + (size_t)(m0 + ml) * K + (half << 3);
  // B fragments: contiguous 32B per lane inside each packed tile
  const _Float16* bbase = Bp + (size_t)(n0 >> 4) * 512 + lane * 16;

  v8f acc0 = {}, acc1 = {}, acc2 = {}, acc3 = {};
  for (int k0 = 0; k0 < K; k0 += 32) {
    v8h alo = *(const v8h*)(arow + k0);
    v8h ahi = *(const v8h*)(arow + k0 + 16);
    v16h a = cat8(alo, ahi);
    const _Float16* bt = bbase + (size_t)(k0 >> 5) * ntile16 * 512;
    v16h b0 = *(const v16h*)(bt);
    v16h b1 = *(const v16h*)(bt + 512);
    v16h b2 = *(const v16h*)(bt + 1024);
    v16h b3 = *(const v16h*)(bt + 1536);
    acc0 = wmma_f16(a, b0, acc0);
    acc1 = wmma_f16(a, b1, acc1);
    acc2 = wmma_f16(a, b2, acc2);
    acc3 = wmma_f16(a, b3, acc3);
  }

  // C/D layout: VGPR r, lanes 0-15 -> M=r, lanes 16-31 -> M=8+r; N = lane&15
  float* crow = C + (size_t)(m0 + (half << 3)) * N + n0 + ml;
#pragma unroll
  for (int r = 0; r < 8; ++r) {
    float* p = crow + (size_t)r * N;
    p[0]  = acc0[r];
    p[16] = acc1[r];
    p[32] = acc2[r];
    p[48] = acc3[r];
  }
}

// ---------------------------------------------------------------------------
// Launch
// ---------------------------------------------------------------------------
extern "C" void kernel_launch(void* const* d_in, const int* in_sizes, int n_in,
                              void* d_out, int out_size, void* d_ws, size_t ws_size,
                              hipStream_t stream) {
  const float*     x    = (const float*)d_in[0];
  const long long* ei   = (const long long*)d_in[1];   // int64 [2, E]
  const float*     W1   = (const float*)d_in[2];
  const float*     b1   = (const float*)d_in[3];
  const float*     Wmu  = (const float*)d_in[4];
  const float*     bmu  = (const float*)d_in[5];
  const float*     Wls  = (const float*)d_in[6];
  const float*     bls  = (const float*)d_in[7];

  const int Cin = 512, H = 256, Co = 128;
  const int N = in_sizes[0] / Cin;   // 50000
  const int E = in_sizes[1] / 2;     // 800000
  const long long* srcv = ei;
  const long long* dstv = ei + E;
  float* out = (float*)d_out;        // [2, N, Co] : mu then logstd

  // ---- workspace carve-up (256B aligned) ----
  char* ws = (char*)d_ws;
  size_t off = 0;
  auto take = [&](size_t bytes) -> char* {
    char* p = ws + off;
    off += (bytes + 255) & ~(size_t)255;
    return p;
  };
  float*    deg  = (float*)take((size_t)N * 4);
  _Float16* xh   = (_Float16*)take((size_t)N * Cin * 2);  // x f16; reused as h f16
  float*    xw1  = (float*)take((size_t)N * H * 4);       // x @ W1
  float*    hacc = (float*)take((size_t)N * H * 4);       // agg accum; reused for xw_mu/xw_ls
  _Float16* W1p  = (_Float16*)take((size_t)Cin * H * 2);
  _Float16* Wmup = (_Float16*)take((size_t)H * Co * 2);
  _Float16* Wlsp = (_Float16*)take((size_t)H * Co * 2);
  _Float16* hf16 = xh;
  float*    xwmu = hacc;
  float*    xwls = hacc + (size_t)N * Co;

  const dim3 blk(256);
  const int mblocks = ((N + 15) / 16 + 7) / 8;  // wave-tiles of 16 rows, 8 per block

  // 1) symmetric-normalization coefficients: deg = 1 (self loop) + count(dst); dinv = rsqrt
  fill_f32<<<256, blk, 0, stream>>>(deg, 1.0f, (size_t)N);
  deg_accum<<<1024, blk, 0, stream>>>(dstv, deg, E);
  rsqrt_inplace<<<256, blk, 0, stream>>>(deg, (size_t)N);  // deg now holds dinv

  // 2) precision conversion / weight packing
  f32_to_f16<<<2048, blk, 0, stream>>>(x, xh, (size_t)N * Cin);
  pack_b_f16<<<256, blk, 0, stream>>>(W1, W1p, Cin, H);
  pack_b_f16<<<64, blk, 0, stream>>>(Wmu, Wmup, H, Co);
  pack_b_f16<<<64, blk, 0, stream>>>(Wls, Wlsp, H, Co);

  // 3) layer 1: xw1 = x @ W1  (WMMA), then normalized scatter-add, bias+ReLU -> h (f16)
  gemm16_wmma<<<dim3(mblocks, H / 64), blk, 0, stream>>>(xh, W1p, xw1, N, H, Cin);
  fill_f32<<<2048, blk, 0, stream>>>(hacc, 0.0f, (size_t)N * H);
  agg_scatter<<<8192, blk, 0, stream>>>(xw1, srcv, dstv, deg, hacc, E, N, 8);
  relu_bias_to_f16<<<2048, blk, 0, stream>>>(hacc, b1, hf16, (size_t)N * H, H - 1);

  // 4) mu / logstd heads: GEMM (WMMA) then scatter into d_out, bias in place
  gemm16_wmma<<<dim3(mblocks, Co / 64), blk, 0, stream>>>(hf16, Wmup, xwmu, N, Co, H);
  gemm16_wmma<<<dim3(mblocks, Co / 64), blk, 0, stream>>>(hf16, Wlsp, xwls, N, Co, H);

  fill_f32<<<2048, blk, 0, stream>>>(out, 0.0f, (size_t)2 * N * Co);
  agg_scatter<<<8192, blk, 0, stream>>>(xwmu, srcv, dstv, deg, out, E, N, 7);
  agg_scatter<<<8192, blk, 0, stream>>>(xwls, srcv, dstv, deg, out + (size_t)N * Co, E, N, 7);
  bias_add_inplace<<<1024, blk, 0, stream>>>(out, bmu, (size_t)N * Co, Co - 1);
  bias_add_inplace<<<1024, blk, 0, stream>>>(out + (size_t)N * Co, bls, (size_t)N * Co, Co - 1);
}
